// RNN_71743133712689
// MI455X (gfx1250) — matrix operand order repaired and checked by reading, not compiled, not run on hardware
//
#include <hip/hip_runtime.h>
#include <hip/hip_bf16.h>
#include <math.h>

typedef __attribute__((ext_vector_type(16))) __bf16 v16bf;
typedef __attribute__((ext_vector_type(8)))  __bf16 v8bf;
typedef __attribute__((ext_vector_type(8)))  float  v8f;
typedef __attribute__((ext_vector_type(4)))  float  v4f;

static constexpr int Bn = 64, Tn = 1024, In = 256, Hn = 512, On = 256;

__device__ __forceinline__ v16bf cat16(v8bf lo, v8bf hi) {
    return __builtin_shufflevector(lo, hi, 0,1,2,3,4,5,6,7,8,9,10,11,12,13,14,15);
}

// Convert 8 contiguous f32 -> v8bf (two 16B f32 loads, in-register cvt)
__device__ __forceinline__ v8bf cvt8(const float* __restrict__ p) {
    v4f a = *(const v4f*)p;
    v4f b = *(const v4f*)(p + 4);
    v8bf r;
    r[0] = (__bf16)a[0]; r[1] = (__bf16)a[1]; r[2] = (__bf16)a[2]; r[3] = (__bf16)a[3];
    r[4] = (__bf16)b[0]; r[5] = (__bf16)b[1]; r[6] = (__bf16)b[2]; r[7] = (__bf16)b[3];
    return r;
}

__device__ __forceinline__ v8f wmma_bf16(v16bf a, v16bf b, v8f c) {
    // 8 args: (neg_a, A, neg_b, B, c_mod, C, reuse_a, reuse_b)
    return __builtin_amdgcn_wmma_f32_16x16x32_bf16(false, a, false, b, (short)0, c,
                                                   false, false);
}

// ---------------------------------------------------------------------------
// Kernel 0: Wh f32 -> bf16 (streamed once; lives in L2 for the recurrence)
// ---------------------------------------------------------------------------
__global__ void cvt_wh_kernel(const float* __restrict__ Wh, __bf16* __restrict__ Whb, int n) {
    for (int i = blockIdx.x * blockDim.x + threadIdx.x; i < n; i += gridDim.x * blockDim.x)
        Whb[i] = (__bf16)Wh[i];
}

// ---------------------------------------------------------------------------
// Kernel 1: xproj[t][b][h] = x[b,t,:] @ Wi^T + bi + bh   (one wave per 16x16 tile)
// M = B*T = 65536 (4096 tiles), N = H = 512 (32 tiles), K = I = 256 (8 chunks)
// ---------------------------------------------------------------------------
__global__ void __launch_bounds__(256)
xproj_kernel(const float* __restrict__ x, const float* __restrict__ Wi,
             const float* __restrict__ bi, const float* __restrict__ bh,
             float* __restrict__ xproj) {
    const int lane = threadIdx.x & 31;
    const int gid  = blockIdx.x * (blockDim.x >> 5) + (threadIdx.x >> 5);
    const int mt   = gid >> 5;           // 0..4095
    const int nt   = gid & 31;           // 0..31
    const int half = lane >> 4;          // 0/1 (K-run selector)
    const int l15  = lane & 15;

    const int rowA = mt * 16 + l15;      // flat row in [B*T]  (x is [B,T,I] row-major)
    const int n    = nt * 16 + l15;      // output column in H

    const float* __restrict__ xrow = x  + (size_t)rowA * In;
    const float* __restrict__ wrow = Wi + (size_t)n    * In;

    v8f acc = {};
    #pragma unroll
    for (int kc = 0; kc < In / 32; ++kc) {
        const int k0 = kc * 32;
        const int kA = k0 + half * 16;   // A: lanes 0-15 K 0..15, lanes 16-31 K 16..31
        v16bf a = cat16(cvt8(xrow + kA), cvt8(xrow + kA + 8));
        const int kB = k0 + half * 16;   // B: lanes 0-15 K 0..15, lanes 16-31 K 16..31
        v16bf b = cat16(cvt8(wrow + kB), cvt8(wrow + kB + 8));
        acc = wmma_bf16(a, b, acc);
    }

    const float bias = bi[n] + bh[n];
    #pragma unroll
    for (int r = 0; r < 8; ++r) {        // C/D: VGPR r -> M = r + 8*half, N = l15
        const int gm = mt * 16 + half * 8 + r;
        const int bb = gm >> 10;         // T = 1024
        const int tt = gm & (Tn - 1);
        xproj[(size_t)tt * (Bn * Hn) + (size_t)bb * Hn + n] = acc[r] + bias;
    }
}

// ---------------------------------------------------------------------------
// Kernel 2: recurrence + output projection.
// 4 workgroups x 512 threads (16 waves). WG owns 16 batch rows; h state (bf16)
// ping-pongs in LDS; per step each wave computes 2 N-tiles over K=512.
// ---------------------------------------------------------------------------
__global__ void __launch_bounds__(512)
rnn_kernel(const float* __restrict__ xproj, const __bf16* __restrict__ Whb,
           const float* __restrict__ Wo, const float* __restrict__ bo,
           float* __restrict__ out) {
    __shared__ __bf16 hbuf[2][16][Hn];   // 2 x 16KB

    const int lane = threadIdx.x & 31;
    const int w    = threadIdx.x >> 5;   // wave 0..15
    const int half = lane >> 4;
    const int l15  = lane & 15;
    const int m0   = blockIdx.x * 16;    // batch-row base for this WG

    // h0 = 0
    for (int i = threadIdx.x; i < 16 * Hn; i += blockDim.x)
        (&hbuf[0][0][0])[i] = (__bf16)0.0f;
    __syncthreads();

    const int n0a = w * 32;
    const int na  = n0a + l15;           // N columns of this wave's two tiles
    const int nb  = n0a + 16 + l15;
    const __bf16* __restrict__ wra = Whb + (size_t)na * Hn;
    const __bf16* __restrict__ wrb = Whb + (size_t)nb * Hn;

    for (int t = 0; t < Tn; ++t) {
        const int cur = t & 1, nxt = cur ^ 1;
        const float* __restrict__ xp = xproj + (size_t)t * (Bn * Hn);

        // C init from precomputed input projection
        v8f ca, cb;
        #pragma unroll
        for (int r = 0; r < 8; ++r) {
            const int m = m0 + half * 8 + r;
            ca[r] = xp[(size_t)m * Hn + na];
            cb[r] = xp[(size_t)m * Hn + nb];
        }
        if (t + 1 < Tn)  // pull next step's tile toward the WGP (global_prefetch_b8)
            __builtin_prefetch(xproj + (size_t)(t + 1) * (Bn * Hn)
                               + (size_t)(m0 + half * 8) * Hn + na, 0, 0);

        #pragma unroll 4
        for (int kc = 0; kc < Hn / 32; ++kc) {
            const int k0 = kc * 32;
            const int kA = k0 + half * 16;
            v16bf a = cat16(*(const v8bf*)&hbuf[cur][l15][kA],
                            *(const v8bf*)&hbuf[cur][l15][kA + 8]);
            const int kB = k0 + half * 16;
            v16bf ba = cat16(*(const v8bf*)(wra + kB), *(const v8bf*)(wra + kB + 8));
            v16bf bb = cat16(*(const v8bf*)(wrb + kB), *(const v8bf*)(wrb + kB + 8));
            ca = wmma_bf16(a, ba, ca);   // two independent accumulator chains
            cb = wmma_bf16(a, bb, cb);
        }

        #pragma unroll
        for (int r = 0; r < 8; ++r) {
            const int ml = half * 8 + r;
            hbuf[nxt][ml][na] = (__bf16)tanhf(ca[r]);
            hbuf[nxt][ml][nb] = (__bf16)tanhf(cb[r]);
        }
        __syncthreads();                 // one barrier per step
    }

    // Fused output projection: h_final (in hbuf[0], since T even) @ Wo^T + bo.
    // 16 waves x one 16-col tile = O=256.
    {
        const int n = w * 16 + l15;
        const float* __restrict__ wrow = Wo + (size_t)n * Hn;
        v8f acc = {};
        #pragma unroll
        for (int kc = 0; kc < Hn / 32; ++kc) {
            const int k0 = kc * 32;
            const int kA = k0 + half * 16;
            v16bf a = cat16(*(const v8bf*)&hbuf[0][l15][kA],
                            *(const v8bf*)&hbuf[0][l15][kA + 8]);
            const int kB = k0 + half * 16;
            v16bf b = cat16(cvt8(wrow + kB), cvt8(wrow + kB + 8));
            acc = wmma_bf16(a, b, acc);
        }
        const float bias = bo[n];
        #pragma unroll
        for (int r = 0; r < 8; ++r) {
            const int m = m0 + half * 8 + r;
            out[(size_t)m * On + n] = acc[r] + bias;
        }
    }
}

// ---------------------------------------------------------------------------
extern "C" void kernel_launch(void* const* d_in, const int* in_sizes, int n_in,
                              void* d_out, int out_size, void* d_ws, size_t ws_size,
                              hipStream_t stream) {
    const float* x  = (const float*)d_in[0];
    const float* Wi = (const float*)d_in[1];
    const float* bi = (const float*)d_in[2];
    const float* Wh = (const float*)d_in[3];
    const float* bh = (const float*)d_in[4];
    const float* Wo = (const float*)d_in[5];
    const float* bo = (const float*)d_in[6];
    float* out = (float*)d_out;

    // ws layout: [0, 512KB) Wh in bf16; [512KB, 512KB+128MB) xproj f32 [T][B][H]
    __bf16* Whb  = (__bf16*)d_ws;
    float* xproj = (float*)((char*)d_ws + (size_t)Hn * Hn * sizeof(__bf16));

    cvt_wh_kernel<<<256, 256, 0, stream>>>(Wh, Whb, Hn * Hn);

    // 4096 M-tiles * 32 N-tiles = 131072 waves, 8 waves/block
    xproj_kernel<<<(4096 * 32) / 8, 256, 0, stream>>>(x, Wi, bi, bh, xproj);

    rnn_kernel<<<Bn / 16, 512, 0, stream>>>(xproj, Whb, Wo, bo, out);
}
